// CausalShapedAttention_9182640079156
// MI455X (gfx1250) — matrix operand order repaired and checked by reading, not compiled, not run on hardware
//
#include <hip/hip_runtime.h>
#include <hip/hip_bf16.h>

typedef __attribute__((ext_vector_type(16))) __bf16 v16bf;
typedef __attribute__((ext_vector_type(8)))  __bf16 v8bf;
typedef __attribute__((ext_vector_type(8)))  float  v8f;
typedef __attribute__((ext_vector_type(4)))  unsigned int u32x4;
typedef __attribute__((ext_vector_type(8)))  int i32x8;
typedef __attribute__((ext_vector_type(4)))  int i32x4;

#define BB 4
#define TT 2048
#define CC 1024
#define HH 16
#define HD 64

__device__ __forceinline__ unsigned short f2bf(float f) {
    unsigned int u = __float_as_uint(f);
    unsigned int r = u + 0x7FFFu + ((u >> 16) & 1u);   // round-to-nearest-even
    return (unsigned short)(r >> 16);
}

__device__ __forceinline__ v8f wmma_bf16(v16bf a, v16bf b, v8f c) {
    return __builtin_amdgcn_wmma_f32_16x16x32_bf16(false, a, false, b, (short)0, c, false, false);
}

__device__ __forceinline__ v16bf join8(v8bf lo, v8bf hi) {
    return __builtin_shufflevector(lo, hi, 0,1,2,3,4,5,6,7,8,9,10,11,12,13,14,15);
}

// ---------------------------------------------------------------------------
// TDM: 2-D tile load Global->LDS, bf16 elements (data_size=2B).
// D# packing per CDNA5 ISA 8.3/8.4.
// ---------------------------------------------------------------------------
__device__ __forceinline__ void tdm_load_2d(unsigned lds_off, const void* gaddr,
        unsigned tensor_d0, unsigned tensor_d1,
        unsigned tile_d0, unsigned tile_d1,
        unsigned stride0, unsigned pad_bits)
{
    unsigned long long ga = (unsigned long long)gaddr;
    u32x4 g0;
    g0[0] = 1u;                                          // count=1, user desc
    g0[1] = lds_off;                                     // lds_addr [63:32]
    g0[2] = (unsigned)ga;                                // global_addr lo
    g0[3] = (unsigned)(ga >> 32) | 0x80000000u;          // addr[56:32] | type=2
    i32x8 g1;
    g1[0] = (int)(0x00010000u | pad_bits);               // data_size=2B (+pad cfg)
    g1[1] = (int)(tensor_d0 << 16);                      // tensor_dim0[15:0]
    g1[2] = (int)((tensor_d0 >> 16) | (tensor_d1 << 16));
    g1[3] = (int)((tensor_d1 >> 16) | (tile_d0 << 16));  // tile_dim0
    g1[4] = (int)tile_d1;                                // tile_dim1 (tile_dim2=0)
    g1[5] = (int)stride0;                                // tensor_dim0_stride lo
    g1[6] = 0;
    g1[7] = 0;                                           // dim1_stride unused (2D)
    i32x4 z4 = {0, 0, 0, 0};
#if __clang_major__ >= 23
    i32x8 z8 = {0, 0, 0, 0, 0, 0, 0, 0};
    __builtin_amdgcn_tensor_load_to_lds(g0, g1, z4, z4, z8, 0);
#else
    __builtin_amdgcn_tensor_load_to_lds(g0, g1, z4, z4, 0);
#endif
}

// pad config for V^T tiles: +8 DWORDs after every 32 DWORDs (row 128B -> 160B)
#define VT_PAD_BITS ((1u << 20) | (4u << 22) | (7u << 25))

// ---------------------------------------------------------------------------
// Kernel 0: fp32 -> bf16 streaming convert (packs 8 elems / thread).
// ---------------------------------------------------------------------------
__global__ __launch_bounds__(256)
void cvt_kernel(const float* __restrict__ src, unsigned short* __restrict__ dst, int n)
{
    const int i = (blockIdx.x * 256 + threadIdx.x) * 8;
    if (i >= n) return;
    const float4 a = *(const float4*)(src + i);
    const float4 b = *(const float4*)(src + i + 4);
    uint4 o;
    o.x = (unsigned)f2bf(a.x) | ((unsigned)f2bf(a.y) << 16);
    o.y = (unsigned)f2bf(a.z) | ((unsigned)f2bf(a.w) << 16);
    o.z = (unsigned)f2bf(b.x) | ((unsigned)f2bf(b.y) << 16);
    o.w = (unsigned)f2bf(b.z) | ((unsigned)f2bf(b.w) << 16);
    *(uint4*)(dst + i) = o;
}

// ---------------------------------------------------------------------------
// Kernel 1: V^T producer: vt[b,h][d][t] = bf16(x[b,t,h*64+d]).
// ---------------------------------------------------------------------------
__global__ __launch_bounds__(128)
void vt_kernel(const float* __restrict__ x, unsigned short* __restrict__ vt)
{
    __shared__ __align__(32) unsigned short tile[64][65];
    const int bh = blockIdx.y, b = bh >> 4, h = bh & 15;
    const int t0 = blockIdx.x * 64;
    const int tid = threadIdx.x;
    const int d = tid & 63, tq = tid >> 6;
    for (int i = 0; i < 64; i += 2) {
        const int t = i + tq;
        tile[t][d] = f2bf(x[((size_t)b * TT + t0 + t) * CC + h * HD + d]);
    }
    __syncthreads();
    const int dd = tid >> 1, seg = (tid & 1) * 32;
    unsigned short* dst = vt + ((size_t)bh * HD + dd) * TT + t0 + seg;
    #pragma unroll
    for (int j = 0; j < 32; ++j) dst[j] = tile[seg + j][dd];
}

// ---------------------------------------------------------------------------
// Kernel 2: qk = x @ w^T (M=8192,N=2048,K=1024), bf16 WMMA.
// TDM double-buffered: tile i+1 DMA overlaps tile i compute.
// ---------------------------------------------------------------------------
#define PJ_BUF (64 * 64 * 2)   // elems per buffer (A 64x64 + B 64x64)

__global__ __launch_bounds__(128)
void proj_kernel(const unsigned short* __restrict__ xbf,
                 const unsigned short* __restrict__ wbf,
                 unsigned short* __restrict__ qbf, unsigned short* __restrict__ kbf)
{
    extern __shared__ unsigned short smem[];
    const unsigned lds0 = __builtin_amdgcn_groupstaticsize();

    const int tid  = threadIdx.x;
    const int wid  = tid >> 5;
    const int lane = tid & 31;
    const int hf   = lane >> 4;
    const int lm   = lane & 15;
    const int m0   = blockIdx.y * 64;
    const int n0   = blockIdx.x * 64;

    v8f acc[4] = { {0,0,0,0,0,0,0,0},{0,0,0,0,0,0,0,0},
                   {0,0,0,0,0,0,0,0},{0,0,0,0,0,0,0,0} };

    const int NIT = CC / 64;   // 16 k-steps of 64

    if (wid == 0) {   // prologue: stage tile 0 into buffer 0
        tdm_load_2d(lds0,              xbf + (size_t)m0 * CC, CC, 8192, 64, 64, CC, 0);
        tdm_load_2d(lds0 + 64*64*2,    wbf + (size_t)n0 * CC, CC, 2048, 64, 64, CC, 0);
    }

    for (int it = 0; it < NIT; ++it) {
        const int cur = it & 1;
        if (wid == 0) __builtin_amdgcn_s_wait_tensorcnt(0);
        __syncthreads();                         // tile `it` visible to all waves
        if (wid == 0 && it + 1 < NIT) {          // prefetch tile it+1 -> other buf
            const int k1 = (it + 1) * 64;
            const unsigned ldsn = lds0 + (unsigned)((1 - cur) * PJ_BUF * 2);
            tdm_load_2d(ldsn,            xbf + (size_t)m0 * CC + k1, CC, 8192, 64, 64, CC, 0);
            tdm_load_2d(ldsn + 64*64*2,  wbf + (size_t)n0 * CC + k1, CC, 2048, 64, 64, CC, 0);
        }

        const unsigned short* sA = smem + cur * PJ_BUF;
        const unsigned short* sB = sA + 64 * 64;

        const unsigned short* ap = &sA[(wid * 16 + lm) * 64];
        v16bf a0 = *(const v16bf*)(ap + hf * 16);
        v16bf a1 = *(const v16bf*)(ap + 32 + hf * 16);
        #pragma unroll
        for (int nt = 0; nt < 4; ++nt) {
            const unsigned short* bp = &sB[(nt * 16 + lm) * 64];
            acc[nt] = wmma_bf16(a0, *(const v16bf*)(bp + hf * 16), acc[nt]);
            acc[nt] = wmma_bf16(a1, *(const v16bf*)(bp + 32 + hf * 16), acc[nt]);
        }
        __syncthreads();                         // done reading buf `cur`
    }

    #pragma unroll
    for (int nt = 0; nt < 4; ++nt) {
        #pragma unroll
        for (int r = 0; r < 8; ++r) {
            const int gm = m0 + wid * 16 + r + 8 * hf;
            const int gn = n0 + nt * 16 + lm;
            const int b = gm >> 11, t = gm & (TT - 1);
            unsigned short* dst;
            float scale;
            int c;
            if (gn < CC) { dst = qbf; scale = 0.125f; c = gn; }     // 1/sqrt(64)
            else         { dst = kbf; scale = 1.0f;   c = gn - CC; }
            const int h = c >> 6, d = c & 63;
            dst[(((size_t)(b * HH + h) * TT + t) << 6) + d] = f2bf(acc[nt][r] * scale);
        }
    }
}

// ---------------------------------------------------------------------------
// Kernel 3: aux terms. y = alpha*(Id@v) - gamma*(MC@v).
// Id@v[t] = ((e-1)v[t]+colsum(v))/(e+T-1) ; MC@v[t] = cumsum(v)[t]/(t+1)
// ---------------------------------------------------------------------------
__global__ __launch_bounds__(64)
void aux_kernel(const float* __restrict__ x,
                const float* __restrict__ alpha_p, const float* __restrict__ gamma_p,
                float* __restrict__ y)
{
    const int bh = blockIdx.x, b = bh >> 4, h = bh & 15;
    const int d = threadIdx.x;
    const float alpha = alpha_p[0], gamma = gamma_p[0];
    const float e1   = __expf(1.0f);
    const float invZ = 1.0f / (e1 + (float)(TT - 1));

    const float* xp = x + (size_t)b * TT * CC + h * HD + d;
    float*       yp = y + (size_t)b * TT * CC + h * HD + d;

    float S = 0.0f;
    for (int t = 0; t < TT; ++t) S += xp[(size_t)t * CC];
    float cum = 0.0f;
    for (int t = 0; t < TT; ++t) {
        const float v = xp[(size_t)t * CC];
        cum += v;
        yp[(size_t)t * CC] = alpha * (((e1 - 1.0f) * v + S) * invZ)
                           - gamma * (cum / (float)(t + 1));
    }
}

// ---------------------------------------------------------------------------
// Kernel 4: causal flash attention, y += beta * softmax(Q K^T) V.
// 64 q-rows/WG, 16 per wave; TDM double-buffered K / V^T tiles.
// ---------------------------------------------------------------------------
#define FL_BUF (64 * 64 + 64 * 80)   // elems per buffer (K tile + padded V^T tile)

__global__ __launch_bounds__(128)
void flash_kernel(const unsigned short* __restrict__ qbf,
                  const unsigned short* __restrict__ kbf,
                  const unsigned short* __restrict__ vt,
                  const float* __restrict__ beta_p,
                  float* __restrict__ y)
{
    extern __shared__ unsigned short smem[];
    unsigned short* Psm = smem + 2 * FL_BUF;         // 4 waves x 16 x 64
    const unsigned lds0 = __builtin_amdgcn_groupstaticsize();

    const int tid  = threadIdx.x;
    const int wid  = tid >> 5;
    const int lane = tid & 31;
    const int hf   = lane >> 4;
    const int lm   = lane & 15;
    const int bh   = blockIdx.y, b = bh >> 4, h = bh & 15;
    const int qt   = blockIdx.x;
    const int q0   = qt * 64;
    const float beta = beta_p[0];

    const unsigned short* qbase  = qbf + ((size_t)bh * TT << 6);
    const unsigned short* kbase  = kbf + ((size_t)bh * TT << 6);
    const unsigned short* vtbase = vt + (size_t)bh * HD * TT;

    // Q rows in registers for the whole kernel (A layout, K=64 -> 2 steps)
    const unsigned short* qp = qbase + ((size_t)(q0 + wid * 16 + lm) << 6);
    v16bf qa[2];
    #pragma unroll
    for (int s = 0; s < 2; ++s)
        qa[s] = *(const v16bf*)(qp + 32 * s + hf * 16);

    v8f o[4] = { {0,0,0,0,0,0,0,0},{0,0,0,0,0,0,0,0},
                 {0,0,0,0,0,0,0,0},{0,0,0,0,0,0,0,0} };
    float mrun[8], lrun[8];
    #pragma unroll
    for (int r = 0; r < 8; ++r) { mrun[r] = -1e30f; lrun[r] = 0.0f; }

    if (wid == 0) {   // prologue: stage key-tile 0 into buffer 0
        tdm_load_2d(lds0,            kbase, 64, TT, 64, 64, 64, 0);
        tdm_load_2d(lds0 + 64*64*2,  vtbase, TT, 64, 64, 64, TT, VT_PAD_BITS);
    }

    for (int kt = 0; kt <= qt; ++kt) {
        const int cur = kt & 1;
        if (wid == 0) __builtin_amdgcn_s_wait_tensorcnt(0);
        __syncthreads();                          // tile `kt` visible to all
        if (wid == 0 && kt < qt) {                // prefetch tile kt+1 -> other buf
            const int k1 = (kt + 1) * 64;
            const unsigned ldsn = lds0 + (unsigned)((1 - cur) * FL_BUF * 2);
            tdm_load_2d(ldsn,            kbase + ((size_t)k1 << 6), 64, TT, 64, 64, 64, 0);
            tdm_load_2d(ldsn + 64*64*2,  vtbase + k1, TT, 64, 64, 64, TT, VT_PAD_BITS);
        }

        const unsigned short* Ksm = smem + cur * FL_BUF;
        const unsigned short* Vsm = Ksm + 64 * 64;

        // ---- S = Q K^T  (16 x 64 per wave) ----
        v8f s[4] = { {0,0,0,0,0,0,0,0},{0,0,0,0,0,0,0,0},
                     {0,0,0,0,0,0,0,0},{0,0,0,0,0,0,0,0} };
        #pragma unroll
        for (int nt = 0; nt < 4; ++nt) {
            #pragma unroll
            for (int ks2 = 0; ks2 < 2; ++ks2) {
                const unsigned short* bp = &Ksm[(nt * 16 + lm) * 64 + ks2 * 32 + hf * 16];
                s[nt] = wmma_bf16(qa[ks2], *(const v16bf*)bp, s[nt]);
            }
        }

        // ---- causal mask on the diagonal tile ----
        if (kt == qt) {
            #pragma unroll
            for (int nt = 0; nt < 4; ++nt)
                #pragma unroll
                for (int r = 0; r < 8; ++r)
                    if (nt * 16 + lm > wid * 16 + r + 8 * hf) s[nt][r] = -1e30f;
        }

        // ---- online softmax (rows live in 16-lane halves of the wave) ----
        #pragma unroll
        for (int r = 0; r < 8; ++r) {
            float v = fmaxf(fmaxf(s[0][r], s[1][r]), fmaxf(s[2][r], s[3][r]));
            #pragma unroll
            for (int m = 8; m >= 1; m >>= 1) v = fmaxf(v, __shfl_xor(v, m, 32));
            const float mnew  = fmaxf(mrun[r], v);
            const float scale = __expf(mrun[r] - mnew);
            mrun[r] = mnew;
            lrun[r] *= scale;
            #pragma unroll
            for (int dt = 0; dt < 4; ++dt) o[dt][r] *= scale;
            float rs = 0.0f;
            #pragma unroll
            for (int nt = 0; nt < 4; ++nt) {
                const float p = __expf(s[nt][r] - mnew);
                s[nt][r] = p;
                rs += p;
            }
            #pragma unroll
            for (int m = 8; m >= 1; m >>= 1) rs += __shfl_xor(rs, m, 32);
            lrun[r] += rs;
        }

        // ---- bounce P through per-wave LDS into A layout ----
        {
            unsigned short* pp = &Psm[wid * 16 * 64];
            #pragma unroll
            for (int nt = 0; nt < 4; ++nt)
                #pragma unroll
                for (int r = 0; r < 8; ++r)
                    pp[(r + 8 * hf) * 64 + nt * 16 + lm] = f2bf(s[nt][r]);
        }

        // ---- O += P V ----
        #pragma unroll
        for (int ks2 = 0; ks2 < 2; ++ks2) {
            const unsigned short* ap = &Psm[wid * 16 * 64 + lm * 64 + 32 * ks2];
            v16bf av = *(const v16bf*)(ap + hf * 16);
            #pragma unroll
            for (int dt = 0; dt < 4; ++dt) {
                const unsigned short* bp = &Vsm[(dt * 16 + lm) * 80 + 32 * ks2 + hf * 16];
                o[dt] = wmma_bf16(av, *(const v16bf*)bp, o[dt]);
            }
        }
        __syncthreads();                          // done reading buf `cur`
    }

    // ---- y += beta * O / l ----
    #pragma unroll
    for (int r = 0; r < 8; ++r) {
        const float inv = beta / lrun[r];
        const int gm = q0 + wid * 16 + r + 8 * hf;
        float* yp = y + ((size_t)b * TT + gm) * CC + h * HD;
        #pragma unroll
        for (int dt = 0; dt < 4; ++dt)
            yp[dt * 16 + lm] += o[dt][r] * inv;
    }
}

extern "C" void kernel_launch(void* const* d_in, const int* in_sizes, int n_in,
                              void* d_out, int out_size, void* d_ws, size_t ws_size,
                              hipStream_t stream) {
    const float* x     = (const float*)d_in[0];
    const float* w     = (const float*)d_in[1];
    const float* alpha = (const float*)d_in[2];
    const float* beta  = (const float*)d_in[3];
    const float* gamma = (const float*)d_in[4];
    float* y = (float*)d_out;

    const size_t nx = (size_t)BB * TT * CC;          // 8.4M
    const size_t nw = (size_t)2 * CC * CC;           // 2.1M
    const size_t per = (size_t)BB * HH * TT * HD;    // 8.4M
    unsigned short* xbf = (unsigned short*)d_ws;
    unsigned short* wbf = xbf + nx;
    unsigned short* qbf = wbf + nw;
    unsigned short* kbf = qbf + per;
    unsigned short* vtb = kbf + per;

    cvt_kernel<<<(int)(nx / 8 / 256), 256, 0, stream>>>(x, xbf, (int)nx);
    cvt_kernel<<<(int)(nw / 8 / 256), 256, 0, stream>>>(w, wbf, (int)nw);
    vt_kernel<<<dim3(TT / 64, BB * HH), 128, 0, stream>>>(x, vtb);

    const size_t proj_lds = (size_t)2 * PJ_BUF * sizeof(unsigned short);    // 32 KB
    proj_kernel<<<dim3(2 * CC / 64, BB * TT / 64), 128, proj_lds, stream>>>(xbf, wbf, qbf, kbf);

    aux_kernel<<<BB * HH, HD, 0, stream>>>(x, alpha, gamma, y);

    const size_t flash_lds = (size_t)(2 * FL_BUF + 4 * 16 * 64) * sizeof(unsigned short); // 44 KB
    flash_kernel<<<dim3(TT / 64, BB * HH), 128, flash_lds, stream>>>(qbf, kbf, vtb, beta, y);
}
